// DrugEncoder_15238543966837
// MI455X (gfx1250) — matrix operand order repaired and checked by reading, not compiled, not run on hardware
//
#include <hip/hip_runtime.h>
#include <hip/hip_bf16.h>

typedef __attribute__((ext_vector_type(16))) __bf16 bf16x16;
typedef __attribute__((ext_vector_type(2)))  __bf16 bf16x2;
typedef __attribute__((ext_vector_type(8)))  float  f32x8;

#define BN_SCALE 0.9999950000374997f
#define TAU_INV  2.0f

// ---------------------------------------------------------------------------
// Generic 128-col GEMM:  out[M,128] = epilogue( A[M,K] @ W[K,128] )
// A is f32 (optionally concat(A,A2), per-half stride 128, when A2!=nullptr).
// LDS tiles are stored pre-swizzled into WMMA fragment order so every lane
// loads its whole 16-element bf16 fragment as two ds_load_b128.
// act: 0=none 1=prelu(alpha) 2=tanh 3=elu ; scale/shift = eval-BN (optional)
// ---------------------------------------------------------------------------
__global__ __launch_bounds__(256)
void gemm_k(const float* __restrict__ A, const float* __restrict__ A2,
            int M, int K,
            const float* __restrict__ W, const float* __restrict__ bias,
            const float* __restrict__ scale, const float* __restrict__ shift,
            const float* __restrict__ alpha_p, int act,
            float* __restrict__ out)
{
    // fragment-order tiles: Asf[wave][lane][16], Bsf[ntile][lane][16]
    __shared__ __attribute__((aligned(32))) __bf16 Asf[8 * 32 * 16];
    __shared__ __attribute__((aligned(32))) __bf16 Bsf[8 * 32 * 16];
    const int tid  = threadIdx.x;
    const int wave = tid >> 5;
    const int lane = tid & 31;
    const int half = lane >> 4;
    const int r    = lane & 15;
    const int rowBase = blockIdx.x * 128;

    f32x8 acc[8];
#pragma unroll
    for (int t = 0; t < 8; ++t)
#pragma unroll
        for (int e = 0; e < 8; ++e) acc[t][e] = 0.f;

    for (int k0 = 0; k0 < K; k0 += 32) {
        // ---- stage A chunk (128 rows x 32 k) as bf16 pairs in frag order ----
        for (int j = tid; j < 128 * 16; j += 256) {
            int rr = j >> 4;             // row in tile
            int kk = (j & 15) * 2;       // even k within chunk
            int grow = rowBase + rr, gk = k0 + kk;
            float v0 = 0.f, v1 = 0.f;
            if (grow < M) {
                if (A2) {
                    // k even => pair never straddles the 128 boundary
                    const float* base = (gk < 128) ? (A  + (long)grow * 128 + gk)
                                                   : (A2 + (long)grow * 128 + gk - 128);
                    if (gk + 1 < K)      { float2 v = *(const float2*)base; v0 = v.x; v1 = v.y; }
                    else if (gk < K)     v0 = base[0];
                } else {
                    const float* base = A + (long)grow * K + gk;
                    if (gk + 1 < K)      { float2 v = *(const float2*)base; v0 = v.x; v1 = v.y; }
                    else if (gk < K)     v0 = base[0];
                }
            }
            // invert A-fragment mapping: kk -> (fhalf, p)
            int fhalf, p;
            if (kk < 16) { fhalf = kk >> 3; p = (kk & 7) >> 1; }
            else         { int k2 = kk - 16; fhalf = k2 >> 3; p = 4 + ((k2 & 7) >> 1); }
            int wv = rr >> 4, fr = rr & 15;
            bf16x2 pr; pr[0] = (__bf16)v0; pr[1] = (__bf16)v1;
            *(bf16x2*)&Asf[((wv * 32 + fhalf * 16 + fr) * 16) + p * 2] = pr;
        }
        // ---- stage W chunk (32 k x 128 n) as bf16 pairs in frag order ----
        for (int j = tid; j < 16 * 128; j += 256) {
            int kk = (j >> 7) * 2;       // even k within chunk
            int nn = j & 127;
            int gk = k0 + kk;
            float v0 = (gk < K)     ? W[(long)gk * 128 + nn]       : 0.f;
            float v1 = (gk + 1 < K) ? W[(long)(gk + 1) * 128 + nn] : 0.f;
            int fhalf, p;
            if (kk < 16) { fhalf = kk >> 3; p = (kk & 7) >> 1; }
            else         { int k2 = kk - 16; fhalf = k2 >> 3; p = 4 + ((k2 & 7) >> 1); }
            int nt = nn >> 4, c = nn & 15;
            bf16x2 pr; pr[0] = (__bf16)v0; pr[1] = (__bf16)v1;
            *(bf16x2*)&Bsf[((nt * 32 + fhalf * 16 + c) * 16) + p * 2] = pr;
        }
        __syncthreads();

        bf16x16 af = *(const bf16x16*)&Asf[(wave * 32 + lane) * 16];
#pragma unroll
        for (int nt = 0; nt < 8; ++nt) {
            bf16x16 bfr = *(const bf16x16*)&Bsf[(nt * 32 + lane) * 16];
            acc[nt] = __builtin_amdgcn_wmma_f32_16x16x32_bf16(
                false, af, false, bfr, (short)0, acc[nt], false, false);
        }
        __syncthreads();
    }

    const float alpha = alpha_p ? alpha_p[0] : 0.f;
#pragma unroll
    for (int nt = 0; nt < 8; ++nt) {
#pragma unroll
        for (int j = 0; j < 8; ++j) {
            int grow = rowBase + wave * 16 + j + half * 8;
            int gcol = nt * 16 + r;
            if (grow < M) {
                float v = acc[nt][j];
                if (bias)  v += bias[gcol];
                if (scale) v = v * (scale[gcol] * BN_SCALE) + shift[gcol];
                if (act == 1)      v = (v >= 0.f) ? v : alpha * v;
                else if (act == 2) v = tanhf(v);
                else if (act == 3) v = (v > 0.f) ? v : (__expf(v) - 1.f);
                out[(long)grow * 128 + gcol] = v;
            }
        }
    }
}

// ------------------------------- helpers -----------------------------------
__global__ void fill_f(float* p, long n, float v) {
    long i = (long)blockIdx.x * 256 + threadIdx.x;
    if (i < n) p[i] = v;
}
__global__ void fill_u(unsigned* p, long n, unsigned v) {
    long i = (long)blockIdx.x * 256 + threadIdx.x;
    if (i < n) p[i] = v;
}

__device__ inline unsigned fenc(float f) {
    unsigned u = __float_as_uint(f);
    return u ^ ((u >> 31) ? 0xFFFFFFFFu : 0x80000000u);
}
__device__ inline float fdec(unsigned u) {
    unsigned b = (u & 0x80000000u) ? (u ^ 0x80000000u) : ~u;
    return __uint_as_float(b);
}
__device__ inline float wred32(float v) {
#pragma unroll
    for (int off = 16; off; off >>= 1) v += __shfl_xor(v, off, 32);
    return v;
}

// r1[i]=dot(x_i,w1); r2[i]=dot(x_i,w2)+b2   (GraphConv scalar head, commuted)
__global__ __launch_bounds__(256)
void dot2_k(const float* x, int n, const float* w1, const float* w2,
            const float* b2, float* r1, float* r2)
{
    int node = blockIdx.x * 8 + (threadIdx.x >> 5);
    int lane = threadIdx.x & 31;
    if (node >= n) return;
    const float* row = x + (long)node * 128;
    float s1 = 0.f, s2 = 0.f;
    for (int c = lane; c < 128; c += 32) { float v = row[c]; s1 += v * w1[c]; s2 += v * w2[c]; }
    s1 = wred32(s1); s2 = wred32(s2);
    if (lane == 0) { r1[node] = s1; r2[node] = s2 + (b2 ? b2[0] : 0.f); }
}

__global__ void scat_s(const int* src, const int* dst, int e, const float* xr, float* s) {
    int i = blockIdx.x * 256 + threadIdx.x;
    if (i < e) atomicAdd(&s[dst[i]], xr[src[i]]);
}
__global__ void segmax_k(const float* s, const int* seg, int n, unsigned* m) {
    int i = blockIdx.x * 256 + threadIdx.x;
    if (i < n) atomicMax(&m[seg[i]], fenc(s[i]));
}
__global__ void segexp_k(const float* s, const int* seg, int n, const unsigned* m,
                         float* esum, float* ebuf) {
    int i = blockIdx.x * 256 + threadIdx.x;
    if (i < n) {
        float mv = fdec(m[seg[i]]);
        if ((__float_as_uint(mv) & 0x7F800000u) == 0x7F800000u) mv = 0.f; // !finite
        float e = __expf(s[i] - mv);
        ebuf[i] = e;
        atomicAdd(&esum[seg[i]], e);
    }
}
__global__ __launch_bounds__(256)
void pool_k(const float* x, int n, const int* seg, const float* ebuf,
            const float* esum, float* pooled) {
    int node = blockIdx.x * 8 + (threadIdx.x >> 5);
    int lane = threadIdx.x & 31;
    if (node >= n) return;
    int b = seg[node];
    float sc = ebuf[node] / (esum[b] + 1e-16f);
    const float* row = x + (long)node * 128;
    float* prow = pooled + (long)b * 128;
    for (int c = lane; c < 128; c += 32) atomicAdd(&prow[c], row[c] * sc);
}
__global__ __launch_bounds__(256)
void rowsum_scat_k(const float* x, int n, const int* seg, float* out) {
    int node = blockIdx.x * 8 + (threadIdx.x >> 5);
    int lane = threadIdx.x & 31;
    if (node >= n) return;
    int b = seg[node];
    const float* row = x + (long)node * 128;
    for (int c = lane; c < 128; c += 32) atomicAdd(&out[(long)b * 128 + c], row[c]);
}

__global__ void deg_k(const int* dst, int e, float* deg) {
    int i = blockIdx.x * 256 + threadIdx.x;
    if (i < e) atomicAdd(&deg[dst[i]], 1.f);
}
__global__ void dis_k(float* d, int n) {
    int i = blockIdx.x * 256 + threadIdx.x;
    if (i < n) d[i] = rsqrtf(d[i] + 1.f);
}
__global__ void gcnself_k(const float* xw, const float* dis, const float* b,
                          long total, float* out) {
    long i = (long)blockIdx.x * 256 + threadIdx.x;
    if (i < total) {
        long node = i >> 7; int c = (int)(i & 127);
        float d = dis[node];
        out[i] = xw[i] * d * d + b[c];
    }
}
__global__ __launch_bounds__(256)
void gcnedge_k(const float* xw, const int* src, const int* dst, int e,
               const float* dis, float* out) {
    int ed = blockIdx.x * 8 + (threadIdx.x >> 5);
    int lane = threadIdx.x & 31;
    if (ed >= e) return;
    int s = src[ed], d = dst[ed];
    float coef = dis[s] * dis[d];
    const float* xr = xw + (long)s * 128;
    float* orow = out + (long)d * 128;
    for (int c = lane; c < 128; c += 32) atomicAdd(&orow[c], xr[c] * coef);
}
__global__ void reluadd_k(float* g, const float* x0, long n) {
    long i = (long)blockIdx.x * 256 + threadIdx.x;
    if (i < n) { float v = g[i] + (x0 ? x0[i] : 0.f); g[i] = v > 0.f ? v : 0.f; }
}
__global__ void eacomb_k(const float* hu, const float* hv, const float* eaw,
                         const int* src, const int* dst, long e128, float* ea) {
    long i = (long)blockIdx.x * 256 + threadIdx.x;
    if (i < e128) {
        long ed = i >> 7; int c = (int)(i & 127);
        ea[i] = (hu[(long)src[ed] * 128 + c] + hv[(long)dst[ed] * 128 + c] + eaw[i]) * (1.f / 3.f);
    }
}

__device__ inline float hash_u01(unsigned a) {
    a ^= a >> 16; a *= 0x7feb352dU; a ^= a >> 15; a *= 0x846ca68bU; a ^= a >> 16;
    return (a >> 8) * (1.f / 16777216.f);
}
__global__ __launch_bounds__(256)
void noise_k(const float* x, int n, unsigned salt, float* out) {
    int node = blockIdx.x * 8 + (threadIdx.x >> 5);
    int lane = threadIdx.x & 31;
    if (node >= n) return;
    float r[4]; float ss = 0.f;
#pragma unroll
    for (int q = 0; q < 4; ++q) {
        int c = lane + q * 32;
        r[q] = hash_u01(((unsigned)node * 128u + (unsigned)c) * 2654435761u ^ salt);
        ss += r[q] * r[q];
    }
    ss = wred32(ss);
    float nrm = fmaxf(sqrtf(ss), 1e-12f);
#pragma unroll
    for (int q = 0; q < 4; ++q) {
        int c = lane + q * 32;
        long idx = (long)node * 128 + c;
        float xv = x[idx];
        float sg = (xv > 0.f) ? 1.f : ((xv < 0.f) ? -1.f : 0.f);
        out[idx] = xv + sg * (r[q] / nrm) * 0.2f;
    }
}

__global__ void score_k(const float* g0, const float* g1, const float* g2,
                        const float* a, const float* ab, int nb, float* sc) {
    int b = blockIdx.x * 256 + threadIdx.x;
    if (b >= nb) return;
    float s0 = 0.f, s1 = 0.f, s2 = 0.f;
    long base = (long)b * 128;
    for (int h = 0; h < 128; ++h) {
        s0 += g0[base + h] * a[h * 3 + 0];
        s1 += g1[base + h] * a[h * 3 + 1];
        s2 += g2[base + h] * a[h * 3 + 2];
    }
    s0 += ab[0]; s1 += ab[1]; s2 += ab[2];
    float m = fmaxf(s0, fmaxf(s1, s2));
    float e0 = __expf(s0 - m), e1 = __expf(s1 - m), e2 = __expf(s2 - m);
    float t = e0 + e1 + e2;
    sc[b * 3 + 0] = e0 / t; sc[b * 3 + 1] = e1 / t; sc[b * 3 + 2] = e2 / t;
}
__global__ void comb_k(const float* o0, const float* o1, const float* o2,
                       const int* seg, const float* sc, long n128, float* out) {
    long i = (long)blockIdx.x * 256 + threadIdx.x;
    if (i < n128) {
        long node = i >> 7;
        const float* scb = &sc[(long)seg[node] * 3];
        out[i] = o0[i] * scb[0] + o1[i] * scb[1] + o2[i] * scb[2];
    }
}
__global__ __launch_bounds__(256)
void norml2_k(const float* x, int n, float eps, float* out) {
    int row = blockIdx.x * 8 + (threadIdx.x >> 5);
    int lane = threadIdx.x & 31;
    if (row >= n) return;
    const float* r = x + (long)row * 128;
    float ss = 0.f;
    for (int c = lane; c < 128; c += 32) { float v = r[c]; ss += v * v; }
    ss = wred32(ss);
    float nrm = fmaxf(sqrtf(ss), eps);
    for (int c = lane; c < 128; c += 32) out[(long)row * 128 + c] = r[c] / nrm;
}
__global__ __launch_bounds__(256)
void sim_k(const float* n1, const float* n2, int nb,
           float* dd, float* rsum, float* csum) {
    __shared__ float rows[8][128];
    int wv = threadIdx.x >> 5, lane = threadIdx.x & 31;
    int row = blockIdx.x * 8 + wv;
    for (int c = lane; c < 128; c += 32)
        rows[wv][c] = (row < nb) ? n1[(long)row * 128 + c] : 0.f;
    __syncthreads();
    if (row >= nb) return;
    float rs = 0.f;
    for (int j = lane; j < nb; j += 32) {
        const float* c2 = n2 + (long)j * 128;
        float dv = 0.f;
        for (int k = 0; k < 128; ++k) dv += rows[wv][k] * c2[k];
        float e = __expf(dv * TAU_INV);
        rs += e;
        atomicAdd(&csum[j], e);
        if (j == row) dd[row] = e;
    }
    rs = wred32(rs);
    if (lane == 0) rsum[row] = rs;
}
__global__ void loss_k(const float* d, const float* rs, const float* cs,
                       int nb, float* out) {
    int b = blockIdx.x * 256 + threadIdx.x;
    if (b < nb) {
        float l1 = -logf(d[b] / (rs[b] + 1e-8f) + 1e-8f);
        float l2 = -logf(d[b] / (cs[b] + 1e-8f) + 1e-8f);
        atomicAdd(out, 0.5f * (l1 + l2) / (float)nb);
    }
}
__global__ void cnt_k(const int* seg, int n, float* cnt) {
    int i = blockIdx.x * 256 + threadIdx.x;
    if (i < n) atomicAdd(&cnt[seg[i]], 1.f);
}
__global__ void divcnt_k(float* ng, const float* cnt, long n128) {
    long i = (long)blockIdx.x * 256 + threadIdx.x;
    if (i < n128) ng[i] /= fmaxf(cnt[i >> 7], 1.f);
}

// ---------------------------------------------------------------------------
extern "C" void kernel_launch(void* const* d_in, const int* in_sizes, int n_in,
                              void* d_out, int out_size, void* d_ws, size_t ws_size,
                              hipStream_t stream)
{
    (void)n_in; (void)ws_size;
    // pytree-flatten order (dict keys alphabetical, lists in order)
    const int*   batch = (const int*)d_in[0];
    const float* eattr = (const float*)d_in[1];
    const int*   eidx  = (const int*)d_in[2];
    const int*   eib   = (const int*)d_in[3];
    const int*   lgidx = (const int*)d_in[4];

    const int N   = in_sizes[0];
    const int E   = in_sizes[3];
    const int ELG = in_sizes[4] / 2;
    const int IN  = in_sizes[72] / N;
    const int EIN = in_sizes[1] / E;
    const int NB  = (out_size - 2) / 128;

    const int *src = eidx, *dst = eidx + E;
    const int *lsrc = lgidx, *ldst = lgidx + ELG;

    auto F = [&](int i) { return (const float*)d_in[i]; };
    // lg
    const float *lgA_Wrel = F(5), *lgA_Wroot = F(6), *lgA_brel = F(7);
    const float *lgB_a = F(8), *lgB_ab = F(9);
    const float *lgB_aWrel = F(10), *lgB_aWroot = F(11), *lgB_abrel = F(12);
    const float *lgW[3] = {F(13), F(15), F(17)}, *lgb[3] = {F(14), F(16), F(18)};
    const float *lgB_gW = F(19), *lgB_gb = F(20);
    const float *lgC_gW1 = F(21), *lgC_gW2 = F(22), *lgC_gb1 = F(23), *lgC_gb2 = F(24);
    const float *lgC_nW1 = F(25), *lgC_nW2 = F(26), *lgC_nb1 = F(27), *lgC_nb2 = F(28);
    const float *lg_ledge = F(29), *lg_lout = F(30), *lg_lu = F(31), *lg_lv = F(32);
    // ng
    const float *ngA_Wrel = F(33), *ngA_Wroot = F(34), *ngA_brel = F(35);
    const float *ngB_a = F(36), *ngB_ab = F(37);
    const float *ngB_aWrel = F(38), *ngB_aWroot = F(39), *ngB_abrel = F(40);
    const float *ngW[3] = {F(41), F(43), F(45)}, *ngb[3] = {F(42), F(44), F(46)};
    const float *ngB_gW = F(47), *ngB_gb = F(48);
    const float *ngC_gW1 = F(49), *ngC_gW2 = F(50), *ngC_gb1 = F(51), *ngC_gb2 = F(52);
    const float *ngC_nW1 = F(53), *ngC_nW2 = F(54), *ngC_nb1 = F(55), *ngC_nb2 = F(56);
    const float *ng_lout = F(57);
    const float *top_W = F(58), *top_b = F(59);
    const float *xmW1 = F(60), *xmW2 = F(61), *xmW3 = F(62);
    const float *xmb1 = F(63), *xmb2 = F(64), *xmb3 = F(65);
    const float *bn1b = F(66), *bn1g = F(67), *bn2b = F(68), *bn2g = F(69);
    const float *p1 = F(70), *p2 = F(71);
    const float *x = F(72);

    // ---- workspace bump allocator ----
    char* wsp = (char*)d_ws; size_t off = 0;
    auto alloc = [&](size_t nf) -> float* {
        float* p = (float*)(wsp + off);
        off += ((nf * 4 + 255) / 256) * 256;
        return p;
    };
    float* h     = alloc((size_t)N * 128);
    float* ea    = alloc((size_t)E * 128);
    float* nea   = alloc((size_t)E * 128);
    float* xw    = alloc((size_t)E * 128);
    float* outs0 = alloc((size_t)E * 128);
    float* outs1 = alloc((size_t)E * 128);
    float* outs2 = alloc((size_t)E * 128);
    float* obuf  = alloc((size_t)E * 128);
    float* s_sc  = alloc((size_t)E);
    float* xrel  = alloc((size_t)E);
    float* ebuf  = alloc((size_t)E);
    float* disE  = alloc((size_t)E);
    float* disN  = alloc((size_t)N);
    unsigned* segm = (unsigned*)alloc((size_t)NB);
    float* segs  = alloc((size_t)NB);
    float* pool  = alloc((size_t)NB * 128);
    float* gout0 = alloc((size_t)NB * 128);
    float* gout1 = alloc((size_t)NB * 128);
    float* gout2 = alloc((size_t)NB * 128);
    float* score = alloc((size_t)NB * 3);
    float* zb1   = alloc((size_t)NB * 128);
    float* zb2   = alloc((size_t)NB * 128);
    float* t1    = alloc((size_t)NB * 128);
    float* t2    = alloc((size_t)NB * 128);
    float* n1b   = alloc((size_t)NB * 128);
    float* n2b   = alloc((size_t)NB * 128);
    float* e_g   = alloc((size_t)NB * 128);
    float* n_g   = alloc((size_t)NB * 128);
    float* dd    = alloc((size_t)NB);
    float* rsum  = alloc((size_t)NB);
    float* csum  = alloc((size_t)NB);
    float* cntb  = alloc((size_t)NB);

    auto gb = [](long n) { return (unsigned)((n + 255) / 256); };
    auto fillf = [&](float* p, long n, float v) {
        fill_f<<<gb(n), 256, 0, stream>>>(p, n, v);
    };
    auto gemm = [&](const float* A, const float* A2, int M, int K, const float* W,
                    const float* bias, const float* sc, const float* sh,
                    const float* al, int act, float* out) {
        gemm_k<<<(M + 127) / 128, 256, 0, stream>>>(A, A2, M, K, W, bias, sc, sh, al, act, out);
    };
    auto att_pool = [&](const float* xin, int n, const int* s, const int* d, int ec,
                        const int* seg, const float* Wrel, const float* Wroot,
                        const float* brel, float* pooled) {
        fillf(pooled, (long)NB * 128, 0.f);
        dot2_k<<<(n + 7) / 8, 256, 0, stream>>>(xin, n, Wrel, Wroot, brel, xrel, s_sc);
        scat_s<<<gb(ec), 256, 0, stream>>>(s, d, ec, xrel, s_sc);
        fill_u<<<gb(NB), 256, 0, stream>>>(segm, NB, 0x007FFFFFu); // enc(-inf)
        segmax_k<<<gb(n), 256, 0, stream>>>(s_sc, seg, n, segm);
        fillf(segs, NB, 0.f);
        segexp_k<<<gb(n), 256, 0, stream>>>(s_sc, seg, n, segm, segs, ebuf);
        pool_k<<<(n + 7) / 8, 256, 0, stream>>>(xin, n, seg, ebuf, segs, pooled);
    };
    auto gcn = [&](const float* xin, int n, const int* s, const int* d, int ec,
                   const float* dis, const float* W, const float* b, float* out) {
        gemm(xin, nullptr, n, 128, W, nullptr, nullptr, nullptr, nullptr, 0, xw);
        gcnself_k<<<gb((long)n * 128), 256, 0, stream>>>(xw, dis, b, (long)n * 128, out);
        gcnedge_k<<<(ec + 7) / 8, 256, 0, stream>>>(xw, s, d, ec, dis, out);
    };
    auto run_block = [&](const float* x0, int n, const int* s, const int* d, int ec,
                         const int* seg, const float* dis,
                         const float* const* layW, const float* const* layB,
                         const float* aWrel, const float* aWroot, const float* abrel,
                         const float* gW, const float* gbias,
                         const float* aP, const float* aB, int residual, float* target) {
        float* outs[3] = {outs0, outs1, outs2};
        float* gouts[3] = {gout0, gout1, gout2};
        for (int l = 0; l < 3; ++l) {
            const float* gin = residual ? (l == 0 ? x0 : outs[l - 1]) : x0;
            gcn(gin, n, s, d, ec, dis, layW[l], layB[l], outs[l]);
            reluadd_k<<<gb((long)n * 128), 256, 0, stream>>>(outs[l], residual ? x0 : nullptr,
                                                             (long)n * 128);
            att_pool(outs[l], n, s, d, ec, seg, aWrel, aWroot, abrel, pool);
            gemm(pool, nullptr, NB, 128, gW, gbias, nullptr, nullptr, nullptr, 2, gouts[l]);
        }
        score_k<<<gb(NB), 256, 0, stream>>>(gout0, gout1, gout2, aP, aB, NB, score);
        comb_k<<<gb((long)n * 128), 256, 0, stream>>>(outs0, outs1, outs2, seg, score,
                                                      (long)n * 128, target);
    };
    auto contrast = [&](const float* z1, const float* z2,
                        const float* W1g, const float* b1g, const float* W2g, const float* b2g,
                        const float* W1n, const float* b1n, const float* W2n, const float* b2n,
                        float* lossp) {
        gemm(z1, nullptr, NB, 128, W1g, b1g, nullptr, nullptr, nullptr, 3, t1);
        gemm(t1, nullptr, NB, 128, W2g, b2g, nullptr, nullptr, nullptr, 0, t2);
        norml2_k<<<(NB + 7) / 8, 256, 0, stream>>>(t2, NB, 1e-8f, n1b);
        gemm(z2, nullptr, NB, 128, W1n, b1n, nullptr, nullptr, nullptr, 3, t1);
        gemm(t1, nullptr, NB, 128, W2n, b2n, nullptr, nullptr, nullptr, 0, t2);
        norml2_k<<<(NB + 7) / 8, 256, 0, stream>>>(t2, NB, 1e-8f, n2b);
        fillf(csum, NB, 0.f);
        fillf(lossp, 1, 0.f);
        sim_k<<<(NB + 7) / 8, 256, 0, stream>>>(n1b, n2b, NB, dd, rsum, csum);
        loss_k<<<gb(NB), 256, 0, stream>>>(dd, rsum, csum, NB, lossp);
    };

    float* g_out    = (float*)d_out;
    float* n_loss_p = g_out + (long)NB * 128;
    float* e_loss_p = n_loss_p + 1;

    // ---- x_mlp: prelu(xW1+b1) -> bn(·W2+b2)->prelu -> bn(·W3+b3) ----
    gemm(x, nullptr, N, IN, xmW1, xmb1, nullptr, nullptr, p1, 1, outs0);
    gemm(outs0, nullptr, N, 128, xmW2, xmb2, bn1g, bn1b, p2, 1, outs1);
    gemm(outs1, nullptr, N, 128, xmW3, xmb3, bn2g, bn2b, nullptr, 0, h);

    // ---- degree / symmetric-norm factors (reused across all layers) ----
    fillf(disE, E, 0.f);
    deg_k<<<gb(ELG), 256, 0, stream>>>(ldst, ELG, disE);
    dis_k<<<gb(E), 256, 0, stream>>>(disE, E);
    fillf(disN, N, 0.f);
    deg_k<<<gb(E), 256, 0, stream>>>(dst, E, disN);
    dis_k<<<gb(N), 256, 0, stream>>>(disN, N);

    // ---- line-graph encoder ----
    float* hu = outs0;
    float* hv = outs0 + (size_t)N * 128;
    float* eaw = xw;
    gemm(h, nullptr, N, 128, lg_lu, nullptr, nullptr, nullptr, nullptr, 0, hu);
    gemm(h, nullptr, N, 128, lg_lv, nullptr, nullptr, nullptr, nullptr, 0, hv);
    gemm(eattr, nullptr, E, EIN, lg_ledge, nullptr, nullptr, nullptr, nullptr, 0, eaw);
    eacomb_k<<<gb((long)E * 128), 256, 0, stream>>>(hu, hv, eaw, src, dst, (long)E * 128, ea);
    noise_k<<<(E + 7) / 8, 256, 0, stream>>>(ea, E, 0x9E3779B9u, nea);

    run_block(ea, E, lsrc, ldst, ELG, eib, disE, lgW, lgb,
              lgB_aWrel, lgB_aWroot, lgB_abrel, lgB_gW, lgB_gb, lgB_a, lgB_ab, 1, obuf);
    run_block(nea, E, lsrc, ldst, ELG, eib, disE, lgW, lgb,
              lgB_aWrel, lgB_aWroot, lgB_abrel, lgB_gW, lgB_gb, lgB_a, lgB_ab, 1, outs0);
    float* no = outs0;
    att_pool(obuf, E, lsrc, ldst, ELG, eib, lgA_Wrel, lgA_Wroot, lgA_brel, zb1);
    att_pool(no,   E, lsrc, ldst, ELG, eib, lgA_Wrel, lgA_Wroot, lgA_brel, zb2);
    contrast(zb1, zb2, lgC_gW1, lgC_gb1, lgC_gW2, lgC_gb2,
             lgC_nW1, lgC_nb1, lgC_nW2, lgC_nb2, e_loss_p);
    gemm(obuf, no, E, 256, lg_lout, nullptr, nullptr, nullptr, nullptr, 0, xw); // oe
    fillf(e_g, (long)NB * 128, 0.f);
    rowsum_scat_k<<<(E + 7) / 8, 256, 0, stream>>>(xw, E, eib, e_g);

    // ---- node-graph encoder ----
    noise_k<<<(N + 7) / 8, 256, 0, stream>>>(h, N, 0x85EBCA6Bu, nea); // nh
    run_block(h, N, src, dst, E, batch, disN, ngW, ngb,
              ngB_aWrel, ngB_aWroot, ngB_abrel, ngB_gW, ngB_gb, ngB_a, ngB_ab, 0, obuf);
    run_block(nea, N, src, dst, E, batch, disN, ngW, ngb,
              ngB_aWrel, ngB_aWroot, ngB_abrel, ngB_gW, ngB_gb, ngB_a, ngB_ab, 0, outs0);
    att_pool(obuf,  N, src, dst, E, batch, ngA_Wrel, ngA_Wroot, ngA_brel, zb1);
    att_pool(outs0, N, src, dst, E, batch, ngA_Wrel, ngA_Wroot, ngA_brel, zb2);
    contrast(zb1, zb2, ngC_gW1, ngC_gb1, ngC_gW2, ngC_gb2,
             ngC_nW1, ngC_nb1, ngC_nW2, ngC_nb2, n_loss_p);
    gemm(obuf, outs0, N, 256, ng_lout, nullptr, nullptr, nullptr, nullptr, 0, xw); // on
    fillf(cntb, NB, 0.f);
    cnt_k<<<gb(N), 256, 0, stream>>>(batch, N, cntb);
    fillf(n_g, (long)NB * 128, 0.f);
    rowsum_scat_k<<<(N + 7) / 8, 256, 0, stream>>>(xw, N, batch, n_g);
    divcnt_k<<<gb((long)NB * 128), 256, 0, stream>>>(n_g, cntb, (long)NB * 128);

    // ---- top projection: concat(e_g, n_g) @ top.W + top.b -> d_out ----
    gemm(e_g, n_g, NB, 256, top_W, top_b, nullptr, nullptr, nullptr, 0, g_out);
}
